// ContentPlanner_841813590461
// MI455X (gfx1250) — matrix operand order repaired and checked by reading, not compiled, not run on hardware
//
#include <hip/hip_runtime.h>
#include <hip/hip_bf16.h>

typedef _Float16 half_t;
typedef _Float16 v16h __attribute__((ext_vector_type(16)));
typedef _Float16 v8h  __attribute__((ext_vector_type(8)));
typedef float    v8f  __attribute__((ext_vector_type(8)));

#define BB 32     // batch
#define LL 700    // records / max_len
#define FF 4      // feats per record
#define HH 600    // hidden
#define H2 1200
#define H4 2400
#define ALD 704   // padded attention row stride (halves) -> 16B-aligned rows

// GEMM tiling: 128x128 block, 8 wave32 waves (4Mx2N), 32x64 per wave
#define TM 128
#define TN 128
#define TK 32
#define LDSS 40   // padded LDS stride in halves (80B rows, conflict-light b128 reads)

enum { AM_F16 = 0, AM_GATHER = 1, AM_CONCAT = 2 };
enum { BM_KN = 0, BM_NK = 1 };
enum { EP_RELU = 0, EP_BIASF16 = 1, EP_F32 = 2, EP_F16 = 3, EP_SIGGATE = 4 };

struct GemmP {
  const half_t* Ah;
  const half_t* A2h;
  const int*    records;
  const float*  emb;
  const half_t* Bh;
  const float*  bias;
  float*        outF;
  half_t*       outH;
  const float*  gateMul;
  int M, N, K;
  int lda, ldb, ldc;
  long strideA, strideB, strideC;
};

__device__ inline v8h zero_v8h() {
  v8h v;
#pragma unroll
  for (int i = 0; i < 8; ++i) v[i] = (half_t)0;
  return v;
}

// gfx1250 async global->LDS copy (ASYNCcnt-tracked, bypasses VGPRs).
// GV mode: LDS address comes from the vdst VGPR, global addr = 64-bit VGPR pair.
__device__ inline void async_ld_b128(const half_t* lds_dst, const half_t* gsrc) {
  const unsigned           ldsoff = (unsigned)(unsigned long long)(size_t)lds_dst;
  const unsigned long long ga     = (unsigned long long)(size_t)gsrc;
  asm volatile("global_load_async_to_lds_b128 %0, %1, off"
               :: "v"(ldsoff), "v"(ga)
               : "memory");
}
__device__ inline void wait_async_all() {
  asm volatile("s_wait_asynccnt 0x0" ::: "memory");
}

template <int AMODE, int BMODE, int EMODE>
__global__ void __launch_bounds__(256) gemm_wmma(GemmP p) {
  __shared__ half_t sA[TM * LDSS];
  __shared__ half_t sB[TN * LDSS];

  const int tid = threadIdx.x;
  const int bz  = blockIdx.z;
  const int m0  = blockIdx.y * TM;
  const int n0  = blockIdx.x * TN;

  const half_t* Ah = p.Ah + (long)bz * p.strideA;
  const half_t* Bh = p.Bh + (long)bz * p.strideB;

  const int wave = tid >> 5;
  const int lane = tid & 31;
  const int wm   = wave >> 1;  // 0..3 -> 32-row slice
  const int wn   = wave & 1;   // 0..1 -> 64-col slice
  const int hsel = lane >> 4;  // 0/1 : lane half (ISA frag layout)
  const int lrow = lane & 15;

  v8f acc[2][4];
#pragma unroll
  for (int i = 0; i < 2; ++i)
#pragma unroll
    for (int j = 0; j < 4; ++j)
#pragma unroll
      for (int e = 0; e < 8; ++e) acc[i][j][e] = 0.f;

  const int ktiles = (p.K + TK - 1) / TK;
  for (int kt = 0; kt < ktiles; ++kt) {
    const int k0 = kt * TK;

    // ---------------- stage A tile (TM x TK) ----------------
    if (AMODE == AM_F16) {
      for (int idx = tid; idx < TM * 4; idx += 256) {
        const int r  = idx >> 2;
        const int c8 = (idx & 3) * 8;
        const int gr = m0 + r;
        const int gk = k0 + c8;
        if (gr < p.M && gk + 8 <= p.K) {
          // async copy straight into LDS, no VGPR round-trip
          async_ld_b128(&sA[r * LDSS + c8], Ah + (long)gr * p.lda + gk);
        } else {
          v8h v = zero_v8h();
          if (gr < p.M) {
            const half_t* src = Ah + (long)gr * p.lda + gk;
#pragma unroll
            for (int i = 0; i < 8; ++i)
              if (gk + i < p.K) v[i] = src[i];
          }
          *(v8h*)(&sA[r * LDSS + c8]) = v;
        }
      }
    } else {
      for (int idx = tid; idx < TM * TK; idx += 256) {
        const int r  = idx >> 5;
        const int c  = idx & 31;
        const int gr = m0 + r;
        const int gk = k0 + c;
        half_t v = (half_t)0;
        if (gr < p.M && gk < p.K) {
          if (AMODE == AM_GATHER) {  // fused emb[records] gather, f32->f16
            const int f   = gk / HH;
            const int h   = gk - f * HH;
            const int tok = p.records[gr * FF + f];
            v = (half_t)p.emb[(long)tok * HH + h];
          } else {  // AM_CONCAT: [emb_relu | emb_att]
            v = (gk < HH) ? p.Ah[(long)gr * HH + gk]
                          : p.A2h[(long)gr * HH + (gk - HH)];
          }
        }
        sA[r * LDSS + c] = v;
      }
    }

    // ---------------- stage B tile ----------------
    if (BMODE == BM_NK) {  // C = A * B^T : B rows are the N dim, K-contiguous
      for (int idx = tid; idx < TN * 4; idx += 256) {
        const int r  = idx >> 2;
        const int c8 = (idx & 3) * 8;
        const int gn = n0 + r;
        const int gk = k0 + c8;
        if (gn < p.N && gk + 8 <= p.K) {
          async_ld_b128(&sB[r * LDSS + c8], Bh + (long)gn * p.ldb + gk);
        } else {
          v8h v = zero_v8h();
          if (gn < p.N) {
            const half_t* src = Bh + (long)gn * p.ldb + gk;
#pragma unroll
            for (int i = 0; i < 8; ++i)
              if (gk + i < p.K) v[i] = src[i];
          }
          *(v8h*)(&sB[r * LDSS + c8]) = v;
        }
      }
    } else {  // BM_KN: transpose-stage row-major KxN weight into Bt[N][K]
      for (int idx = tid; idx < TN * TK; idx += 256) {
        const int n  = idx & 127;
        const int c  = idx >> 7;
        const int gn = n0 + n;
        const int gk = k0 + c;
        half_t v = (half_t)0;
        if (gn < p.N && gk < p.K) v = Bh[(long)gk * p.ldb + gn];
        sB[n * LDSS + c] = v;
      }
    }

    // drain this wave's async LDS writes before signaling the barrier
    if (AMODE == AM_F16 || BMODE == BM_NK) wait_async_all();
    __syncthreads();

    // ---------------- fragments + 8x WMMA ----------------
    v16h afrag[2];
    v16h bfrag[4];
#pragma unroll
    for (int mi = 0; mi < 2; ++mi) {
      // A 16x32 f16 layout: lane row = lrow, kbase = hsel*8, halves [kb,kb+8) and [kb+16,kb+24)
      const half_t* base = &sA[(wm * 32 + mi * 16 + lrow) * LDSS + hsel * 8];
      union { v16h v; v8h h[2]; } u;
      u.h[0] = *(const v8h*)(base);
      u.h[1] = *(const v8h*)(base + 16);
      afrag[mi] = u.v;
    }
#pragma unroll
    for (int ni = 0; ni < 4; ++ni) {
      // B 32x16 f16 layout: lane col = lrow, kbase = hsel*8, halves [kb,kb+8) and [kb+16,kb+24)
      const half_t* base = &sB[(wn * 64 + ni * 16 + lrow) * LDSS + hsel * 8];
      union { v16h v; v8h h[2]; } u;
      u.h[0] = *(const v8h*)(base);
      u.h[1] = *(const v8h*)(base + 16);
      bfrag[ni] = u.v;
    }
#pragma unroll
    for (int mi = 0; mi < 2; ++mi)
#pragma unroll
      for (int ni = 0; ni < 4; ++ni)
        acc[mi][ni] = __builtin_amdgcn_wmma_f32_16x16x32_f16(
            false, afrag[mi], false, bfrag[ni], (short)0, acc[mi][ni], false, false);

    __syncthreads();
  }

  // ---------------- epilogue (C/D layout: row = base + hsel*8 + v, col = lrow) --
#pragma unroll
  for (int mi = 0; mi < 2; ++mi) {
#pragma unroll
    for (int ni = 0; ni < 4; ++ni) {
      const int gn = n0 + wn * 64 + ni * 16 + lrow;
      if (gn >= p.N) continue;
      const int gmb = m0 + wm * 32 + mi * 16 + hsel * 8;
#pragma unroll
      for (int v = 0; v < 8; ++v) {
        const int gm = gmb + v;
        if (gm >= p.M) break;
        const float a  = acc[mi][ni][v];
        const long off = (long)bz * p.strideC + (long)gm * p.ldc + gn;
        if (EMODE == EP_RELU) {
          float r = a + p.bias[gn];
          r = r > 0.f ? r : 0.f;
          p.outF[off] = r;
          p.outH[off] = (half_t)r;
        } else if (EMODE == EP_BIASF16) {
          p.outH[off] = (half_t)(a + p.bias[gn]);
        } else if (EMODE == EP_F32) {
          p.outF[off] = a;
        } else if (EMODE == EP_F16) {
          p.outH[off] = (half_t)a;
        } else {  // EP_SIGGATE: sc = sigmoid(acc+b) * emb_relu
          const float g = 1.f / (1.f + __expf(-(a + p.bias[gn])));
          p.outF[off] = g * p.gateMul[off];
        }
      }
    }
  }
}

// ---------------- reductions (wave32) ----------------
__device__ inline float wave_max(float v) {
#pragma unroll
  for (int o = 16; o > 0; o >>= 1) v = fmaxf(v, __shfl_xor(v, o, 32));
  return v;
}
__device__ inline float wave_sum(float v) {
#pragma unroll
  for (int o = 16; o > 0; o >>= 1) v += __shfl_xor(v, o, 32);
  return v;
}
__device__ inline float block_max(float v, float* red) {
  const int tid = threadIdx.x;
  v = wave_max(v);
  if ((tid & 31) == 0) red[tid >> 5] = v;
  __syncthreads();
  if (tid == 0) {
    float m = red[0];
    for (int i = 1; i < 8; ++i) m = fmaxf(m, red[i]);
    red[0] = m;
  }
  __syncthreads();
  v = red[0];
  __syncthreads();
  return v;
}
__device__ inline float block_sum(float v, float* red) {
  const int tid = threadIdx.x;
  v = wave_sum(v);
  if ((tid & 31) == 0) red[tid >> 5] = v;
  __syncthreads();
  if (tid == 0) {
    float m = 0.f;
    for (int i = 0; i < 8; ++i) m += red[i];
    red[0] = m;
  }
  __syncthreads();
  v = red[0];
  __syncthreads();
  return v;
}

// ---------------- small kernels ----------------
__global__ void __launch_bounds__(256) f32_to_f16(const float* in, half_t* out, int n) {
  const int i = blockIdx.x * 256 + threadIdx.x;
  if (i < n) out[i] = (half_t)in[i];
}

__global__ void __launch_bounds__(256) pad_kernel(const int* records, int* pad) {
  const int i = blockIdx.x * 256 + threadIdx.x;
  if (i < BB * LL) {
    const int* r = records + i * FF;
    int mx = r[0];
    mx = r[1] > mx ? r[1] : mx;
    mx = r[2] > mx ? r[2] : mx;
    mx = r[3] > mx ? r[3] : mx;
    pad[i] = (mx == 0) ? 1 : 0;
  }
}

__global__ void __launch_bounds__(256) softmax_rows(const float* preattn, const int* pad,
                                                    half_t* attnH) {
  __shared__ float red[8];
  const int l   = blockIdx.x;
  const int b   = blockIdx.y;
  const int tid = threadIdx.x;
  const float* row = preattn + ((long)b * LL + l) * LL;
  half_t* orow     = attnH + ((long)b * LL + l) * ALD;
  const int* padb  = pad + b * LL;
  const bool rowPad = padb[l] != 0;

  float mx = -__builtin_inff();
  for (int m = tid; m < LL; m += 256) {
    const bool masked = rowPad || (m == l) || (padb[m] != 0);
    if (!masked) mx = fmaxf(mx, row[m]);
  }
  mx = block_max(mx, red);
  const bool dead = !(mx > -__builtin_inff());

  float s = 0.f;
  for (int m = tid; m < LL; m += 256) {
    const bool masked = dead || rowPad || (m == l) || (padb[m] != 0);
    if (!masked) s += __expf(row[m] - mx);
  }
  s = block_sum(s, red);
  const float inv = (s > 0.f) ? (1.f / s) : 0.f;

  for (int m = tid; m < ALD; m += 256) {
    float o = 0.f;
    if (m < LL) {
      const bool masked = dead || rowPad || (m == l) || (padb[m] != 0);
      if (!masked) o = __expf(row[m] - mx) * inv;
    }
    orow[m] = (half_t)o;
  }
}

__global__ void __launch_bounds__(256) mean_gather(const float* sc, const int* index,
                                                   float* hidden0, float* xsel) {
  const int i = blockIdx.x * 256 + threadIdx.x;  // B*H
  if (i >= BB * HH) return;
  const int b = i / HH;
  const int h = i - b * HH;
  const float* base = sc + (long)b * LL * HH + h;
  float s = 0.f;
  for (int l = 0; l < LL; ++l) s += base[(long)l * HH];
  hidden0[i] = s / (float)LL;
  const int idx = index[b];
  xsel[i] = sc[((long)b * LL + idx) * HH + h];
}

__global__ void __launch_bounds__(256) lstm_gates(const float* x, const float* h0,
                                                  const float* W_ih, const float* W_hh,
                                                  const float* b_ih, const float* b_hh,
                                                  float* gates) {
  const int i = blockIdx.x * 256 + threadIdx.x;  // B*4H
  if (i >= BB * H4) return;
  const int b = i / H4;
  const int j = i - b * H4;
  float s = b_ih[j] + b_hh[j];
  const float* xb = x + b * HH;
  const float* hb = h0 + b * HH;
  for (int k = 0; k < HH; ++k)
    s += xb[k] * W_ih[(long)k * H4 + j] + hb[k] * W_hh[(long)k * H4 + j];
  gates[i] = s;
}

__global__ void __launch_bounds__(256) lstm_cell(const float* gates, float* hidden, float* cell) {
  const int i = blockIdx.x * 256 + threadIdx.x;  // B*H
  if (i >= BB * HH) return;
  const int b = i / HH;
  const int h = i - b * HH;
  const float* g = gates + (long)b * H4;
  const float ig = 1.f / (1.f + __expf(-g[h]));
  const float og = 1.f / (1.f + __expf(-g[3 * HH + h]));
  const float gg = tanhf(g[2 * HH + h]);
  const float c  = ig * gg;  // cell starts at zero -> forget term vanishes
  cell[i]   = c;
  hidden[i] = og * tanhf(c);
}

__global__ void __launch_bounds__(256) logits_kernel(const float* hidden, const float* W_log,
                                                     const float* b_log, const int* pad,
                                                     const int* index, float* out) {
  const int i = blockIdx.x * 256 + threadIdx.x;  // B*L
  if (i >= BB * LL) return;
  const int b = i / LL;
  const int j = i - b * LL;
  float s = b_log[j];
  const float* hb = hidden + b * HH;
  for (int k = 0; k < HH; ++k) s += hb[k] * W_log[(long)k * LL + j];
  const bool masked = (pad[b * LL + j] != 0) || (j == index[b]);
  out[i] = masked ? -__builtin_inff() : s;
}

__global__ void __launch_bounds__(256) logsoftmax_rows(float* out) {
  __shared__ float red[8];
  const int b   = blockIdx.x;
  const int tid = threadIdx.x;
  float* row = out + (long)b * LL;
  float mx = -__builtin_inff();
  for (int m = tid; m < LL; m += 256) mx = fmaxf(mx, row[m]);
  mx = block_max(mx, red);
  float s = 0.f;
  for (int m = tid; m < LL; m += 256) {
    const float v = row[m];
    if (v > -__builtin_inff()) s += __expf(v - mx);
  }
  s = block_sum(s, red);
  const float lse = logf(s);
  for (int m = tid; m < LL; m += 256) row[m] = row[m] - mx - lse;
}

// ---------------- launcher ----------------
extern "C" void kernel_launch(void* const* d_in, const int* in_sizes, int n_in,
                              void* d_out, int out_size, void* d_ws, size_t ws_size,
                              hipStream_t stream) {
  (void)in_sizes; (void)n_in; (void)out_size; (void)ws_size;
  const int*   records = (const int*)  d_in[0];
  const int*   index   = (const int*)  d_in[1];
  const float* emb     = (const float*)d_in[2];
  const float* W_relu  = (const float*)d_in[3];
  const float* b_relu  = (const float*)d_in[4];
  const float* W_lin   = (const float*)d_in[5];
  const float* b_lin   = (const float*)d_in[6];
  const float* W_sig   = (const float*)d_in[7];
  const float* b_sig   = (const float*)d_in[8];
  const float* W_ih    = (const float*)d_in[9];
  const float* W_hh    = (const float*)d_in[10];
  const float* b_ih    = (const float*)d_in[11];
  const float* b_hh    = (const float*)d_in[12];
  const float* W_log   = (const float*)d_in[13];
  const float* b_log   = (const float*)d_in[14];

  float* out_att    = (float*)d_out;                // [B, L]  log-softmax attention
  float* out_hidden = out_att + (size_t)BB * LL;    // [B, H]
  float* out_cell   = out_hidden + (size_t)BB * HH; // [B, H]

  char*  ws  = (char*)d_ws;
  size_t off = 0;
  auto take = [&](size_t bytes) -> char* {
    char* pp = ws + off;
    off += (bytes + 255) & ~(size_t)255;
    return pp;
  };
  half_t* Wrelu_h   = (half_t*)take((size_t)H4 * HH * 2);
  half_t* Wlin_h    = (half_t*)take((size_t)HH * HH * 2);
  half_t* Wsig_h    = (half_t*)take((size_t)H2 * HH * 2);
  half_t* embrelu_h = (half_t*)take((size_t)BB * LL * HH * 2);
  float*  embrelu_f = (float*) take((size_t)BB * LL * HH * 4);
  half_t* emblin_h  = (half_t*)take((size_t)BB * LL * HH * 2);
  float*  preattn   = (float*) take((size_t)BB * LL * LL * 4);
  half_t* attn_h    = (half_t*)take((size_t)BB * LL * ALD * 2);
  int*    pad       = (int*)   take((size_t)BB * LL * 4);
  float*  hidden0   = (float*) take((size_t)BB * HH * 4);
  float*  xsel      = (float*) take((size_t)BB * HH * 4);
  float*  gatesbuf  = (float*) take((size_t)BB * H4 * 4);
  half_t* embatt_h  = emblin_h;  // alias: emb_lin dead after attn GEMM 1
  float*  sc        = preattn;   // alias: pre_attn dead after softmax

  // weight conversions + pad mask
  f32_to_f16<<<(H4 * HH + 255) / 256, 256, 0, stream>>>(W_relu, Wrelu_h, H4 * HH);
  f32_to_f16<<<(HH * HH + 255) / 256, 256, 0, stream>>>(W_lin, Wlin_h, HH * HH);
  f32_to_f16<<<(H2 * HH + 255) / 256, 256, 0, stream>>>(W_sig, Wsig_h, H2 * HH);
  pad_kernel<<<(BB * LL + 255) / 256, 256, 0, stream>>>(records, pad);

  // GEMM1: relu(emb[records] @ W_relu + b)  [22400 x 2400 x 600]
  {
    GemmP p{};
    p.records = records; p.emb = emb; p.Bh = Wrelu_h; p.bias = b_relu;
    p.outF = embrelu_f; p.outH = embrelu_h;
    p.M = BB * LL; p.N = HH; p.K = H4; p.ldb = HH; p.ldc = HH;
    gemm_wmma<AM_GATHER, BM_KN, EP_RELU><<<dim3(5, 175, 1), 256, 0, stream>>>(p);
  }
  // GEMM2: emb_relu @ W_lin + b -> f16
  {
    GemmP p{};
    p.Ah = embrelu_h; p.Bh = Wlin_h; p.bias = b_lin; p.outH = emblin_h;
    p.M = BB * LL; p.N = HH; p.K = HH; p.lda = HH; p.ldb = HH; p.ldc = HH;
    gemm_wmma<AM_F16, BM_KN, EP_BIASF16><<<dim3(5, 175, 1), 256, 0, stream>>>(p);
  }
  // pre_attn[b] = emb_lin[b] @ emb_relu[b]^T  (B rows map straight to WMMA-B layout)
  {
    GemmP p{};
    p.Ah = emblin_h; p.Bh = embrelu_h; p.outF = preattn;
    p.M = LL; p.N = LL; p.K = HH; p.lda = HH; p.ldb = HH; p.ldc = LL;
    p.strideA = (long)LL * HH; p.strideB = (long)LL * HH; p.strideC = (long)LL * LL;
    gemm_wmma<AM_F16, BM_NK, EP_F32><<<dim3(6, 6, BB), 256, 0, stream>>>(p);
  }
  // masked softmax -> f16 attention (stride 704)
  softmax_rows<<<dim3(LL, BB), 256, 0, stream>>>(preattn, pad, attn_h);
  // emb_att[b] = attn[b] @ emb_relu[b]
  {
    GemmP p{};
    p.Ah = attn_h; p.Bh = embrelu_h; p.outH = embatt_h;
    p.M = LL; p.N = HH; p.K = LL; p.lda = ALD; p.ldb = HH; p.ldc = HH;
    p.strideA = (long)LL * ALD; p.strideB = (long)LL * HH; p.strideC = (long)LL * HH;
    gemm_wmma<AM_F16, BM_KN, EP_F16><<<dim3(5, 6, BB), 256, 0, stream>>>(p);
  }
  // sc = sigmoid([emb_relu|emb_att] @ W_sig + b) * emb_relu
  {
    GemmP p{};
    p.Ah = embrelu_h; p.A2h = embatt_h; p.Bh = Wsig_h; p.bias = b_sig;
    p.outF = sc; p.gateMul = embrelu_f;
    p.M = BB * LL; p.N = HH; p.K = H2; p.lda = HH; p.ldb = HH; p.ldc = HH;
    gemm_wmma<AM_CONCAT, BM_KN, EP_SIGGATE><<<dim3(5, 175, 1), 256, 0, stream>>>(p);
  }
  // tail: mean/gather, LSTM step, logits, masked log-softmax
  mean_gather<<<(BB * HH + 255) / 256, 256, 0, stream>>>(sc, index, hidden0, xsel);
  lstm_gates<<<(BB * H4 + 255) / 256, 256, 0, stream>>>(xsel, hidden0, W_ih, W_hh, b_ih, b_hh, gatesbuf);
  lstm_cell<<<(BB * HH + 255) / 256, 256, 0, stream>>>(gatesbuf, out_hidden, out_cell);
  logits_kernel<<<(BB * LL + 255) / 256, 256, 0, stream>>>(out_hidden, W_log, b_log, pad, index, out_att);
  logsoftmax_rows<<<BB, 256, 0, stream>>>(out_att);
}
